// Max_attention_79053168050388
// MI455X (gfx1250) — matrix operand order repaired and checked by reading, not compiled, not run on hardware
//
#include <hip/hip_runtime.h>
#include <math.h>

typedef __attribute__((ext_vector_type(16))) __bf16 v16bf;
typedef __attribute__((ext_vector_type(8)))  __bf16 v8bf;
typedef __attribute__((ext_vector_type(8)))  float  v8f;
typedef __attribute__((ext_vector_type(4)))  float  v4f;

#define LDQKV 18432   // 3*C, row stride of qkv buffer
#define CDIM  6144
#define DHEAD 3072
#define NSEQ  240
#define SLD   256     // padded row stride of softmax matrix (240 -> 256)
#define KC    64      // K-chunk (bf16) staged in LDS per double-buffer slot

// ---------------------------------------------------------------------------
// Fragment k order used in this file (wave32; contiguous, same for A and B):
// A (16x32): lane L -> row L&15; lanes 0-15: K 0..15, lanes 16-31: K 16..31.
// B (32x16): lane L -> col L&15; lanes 0-15: K 0..15, lanes 16-31: K 16..31.
// ---------------------------------------------------------------------------
static __device__ inline v16bf load_a_frag_f32(const float* __restrict__ tile,
                                               int ld, int k0, int lane) {
  int r   = lane & 15;
  int khi = (lane & 16) ? 16 : 0;
  const float* p = tile + (size_t)r * ld + k0 + khi;
  v16bf f;
#pragma unroll
  for (int i = 0; i < 8; ++i) f[i] = (__bf16)p[i];
#pragma unroll
  for (int i = 0; i < 8; ++i) f[8 + i] = (__bf16)p[8 + i];
  return f;
}

// A fragment from an LDS-staged [16 rows][KC] bf16 block: pure ds_load path.
static __device__ inline v16bf load_a_frag_lds_bf(const __bf16* Ablk, int kk, int lane) {
  int r   = lane & 15;
  int khi = (lane & 16) ? 16 : 0;
  const v8bf* p = (const v8bf*)(Ablk + (size_t)r * KC + kk + khi);
  v8bf lo = p[0];        // K khi..khi+7
  v8bf hi = p[1];        // K khi+8..khi+15
  return __builtin_shufflevector(lo, hi, 0, 1, 2, 3, 4, 5, 6, 7,
                                 8, 9, 10, 11, 12, 13, 14, 15);
}

// B operand for D = A @ W^T (cached path): lane's col c = row c of W.
static __device__ inline v16bf load_b_frag_rowmajor(const float* __restrict__ w0,
                                                    int ld, int k0, int lane) {
  int c  = lane & 15;
  int kb = k0 + ((lane & 16) ? 16 : 0);
  const float* p = w0 + (size_t)c * ld + kb;
  v16bf f;
#pragma unroll
  for (int i = 0; i < 16; ++i) f[i] = (__bf16)p[i];
  return f;
}

// B operand, non-temporal: single-use weight stream must not flush L2
// (the reused activations + qkv + S all fit in the 192MB L2).
static __device__ inline v16bf load_b_frag_nt(const float* __restrict__ w0,
                                              int ld, int k0, int lane) {
  int c  = lane & 15;
  int kb = k0 + ((lane & 16) ? 16 : 0);
  const v4f* p = (const v4f*)(w0 + (size_t)c * ld + kb);
  v4f q0 = __builtin_nontemporal_load(p);
  v4f q1 = __builtin_nontemporal_load(p + 1);
  v4f q2 = __builtin_nontemporal_load(p + 2);
  v4f q3 = __builtin_nontemporal_load(p + 3);
  v16bf f;
#pragma unroll
  for (int i = 0; i < 4; ++i) {
    f[i]      = (__bf16)q0[i];
    f[4 + i]  = (__bf16)q1[i];
    f[8 + i]  = (__bf16)q2[i];
    f[12 + i] = (__bf16)q3[i];
  }
  return f;
}

// B operand for D = A @ B with B row-major (K = rows): strided gather, guarded.
static __device__ inline v16bf load_b_frag_strided(const float* __restrict__ B,
                                                   int ldb, int col0, int k0,
                                                   int kmax, int lane) {
  int c  = lane & 15;
  int kb = k0 + ((lane & 16) ? 16 : 0);
  v16bf f;
#pragma unroll
  for (int i = 0; i < 16; ++i) {
    int k  = kb + i;
    float v = (k < kmax) ? B[(size_t)k * ldb + col0 + c] : 0.0f;
    f[i] = (__bf16)v;
  }
  return f;
}

static __device__ inline v8f wmma_bf16(v16bf a, v16bf b, v8f c) {
  return __builtin_amdgcn_wmma_f32_16x16x32_bf16(
      /*neg_a=*/false, a, /*neg_b=*/false, b,
      /*c_mod=*/(short)0, c, /*reuse_a=*/false, /*reuse_b=*/false);
}

// ---------------------------------------------------------------------------
// Async-stage one 240-row x KC-col bf16 A-chunk into LDS.
// 240*64*2 = 30720B = 3840 x 8B; 256 threads x 15 per-lane B64 copies each
// -> every wave issues exactly 15 ASYNCcnt ops per chunk (uniform waits).
// NOT unrolled: keeps only one live address pair so the 15 accumulators of
// the compute loop never get pushed over the 256-VGPR cliff (no spills).
// ---------------------------------------------------------------------------
static __device__ inline void stage_abf_chunk(const __bf16* __restrict__ A,
                                              int lda, int kb, unsigned ldsbuf,
                                              int tid) {
#pragma unroll 1
  for (int s = 0; s < 15; ++s) {
    int q   = tid + s * 256;        // 0..3839
    int row = q >> 4;               // 240 rows, 16 x 8B per row
    int ke  = (q & 15) * 4;         // element offset (4 bf16 = 8B)
    const __bf16* g = A + (size_t)row * lda + kb + ke;
    unsigned loff   = ldsbuf + (unsigned)(row * KC + ke) * 2u;
    asm volatile("global_load_async_to_lds_b64 %0, %1, off"
                 :: "v"(loff), "v"((unsigned long long)(uintptr_t)g)
                 : "memory");
  }
}

// ---------------------------------------------------------------------------
// Weight-stream GEMM: D[m, col] = sum_k A[m,k] * W[col,k] (+ bias[col])
// for ALL 240 rows in one block. Block = 8 waves; each wave owns 16 columns,
// holds one B fragment and reuses it across 15 M-tiles (15 accumulators).
// => every weight byte enters a VGPR exactly once (grid has no M dimension).
// A is bf16, double-buffered in LDS via async copies; W fp32 streamed NT.
// A-fragments software-pipelined by one stage to hide ds_load latency.
// Requires K % KC == 0, M == 240.
// ---------------------------------------------------------------------------
__global__ void gemm_wstream_bf16(const __bf16* __restrict__ A, int lda,
                                  const float* __restrict__ W, int ldw,
                                  float* __restrict__ D, int ldd, int K,
                                  const float* __restrict__ bias) {
  __shared__ __bf16 As[2][NSEQ * KC];   // 2 x 30720 B
  int tid  = threadIdx.x;
  int lane = tid & 31;
  int wave = tid >> 5;
  int col0 = blockIdx.x * 128 + wave * 16;
  const float* Wband = W + (size_t)col0 * ldw;

  typedef __attribute__((address_space(3))) __bf16 lds_bf;
  unsigned lds0 = (unsigned)(uintptr_t)(lds_bf*)&As[0][0];
  unsigned lds1 = (unsigned)(uintptr_t)(lds_bf*)&As[1][0];

  v8f acc[15] = {};
  int nch = K / KC;

  stage_abf_chunk(A, lda, 0, lds0, tid);
  for (int c = 0; c < nch; ++c) {
    int cur = c & 1;
    if (c + 1 < nch) {
      stage_abf_chunk(A, lda, (c + 1) * KC, cur ? lds0 : lds1, tid);
      asm volatile("s_wait_asynccnt 0xf" ::: "memory");  // chunk c landed
    } else {
      asm volatile("s_wait_asynccnt 0x0" ::: "memory");
    }
    __syncthreads();

    const __bf16* Ac = &As[cur][0];
#pragma unroll
    for (int kk = 0; kk < KC; kk += 32) {
      int k0 = c * KC + kk;
      // prefetch the weight stream ahead (speculative, OOB-safe)
      __builtin_prefetch(W + (size_t)(col0 + (lane & 15)) * ldw + k0 + 512, 0, 1);

      v16bf b = load_b_frag_nt(Wband, ldw, k0, lane);
      v16bf a = load_a_frag_lds_bf(Ac, kk, lane);
#pragma unroll
      for (int m = 0; m < 15; ++m) {
        v16bf an = a;
        if (m < 14)  // prefetch next M-tile's A fragment before the WMMA
          an = load_a_frag_lds_bf(Ac + (m + 1) * 16 * KC, kk, lane);
        acc[m] = wmma_bf16(a, b, acc[m]);  // disjoint D, shared B: no hazards
        a = an;
      }
    }
    __syncthreads();   // safe to overwrite buffer 'cur' next time around
  }

  int n  = lane & 15;
  int mb = (lane & 16) ? 8 : 0;
  float bv = bias ? bias[col0 + n] : 0.0f;
#pragma unroll
  for (int m = 0; m < 15; ++m)
#pragma unroll
    for (int i = 0; i < 8; ++i)
      D[(size_t)(m * 16 + mb + i) * ldd + col0 + n] = acc[m][i] + bv;
}

// ---------------------------------------------------------------------------
// fp32 -> bf16 elementwise convert (for the x activations)
// ---------------------------------------------------------------------------
__global__ void f32_to_bf16_kernel(const float* __restrict__ in,
                                   __bf16* __restrict__ out, int n) {
  int i = blockIdx.x * blockDim.x + threadIdx.x;
  if (i < n) out[i] = (__bf16)in[i];
}

// ---------------------------------------------------------------------------
// MaxPool1d(256) over q's feature dim: pool[h, n, j], j in [0,12)
// ---------------------------------------------------------------------------
__global__ void qpool_kernel(const float* __restrict__ qkv, float* __restrict__ pool) {
  int idx = blockIdx.x * blockDim.x + threadIdx.x;
  if (idx >= 2 * NSEQ * 12) return;
  int j = idx % 12;
  int n = (idx / 12) % NSEQ;
  int h = idx / (12 * NSEQ);
  const float* p = qkv + (size_t)n * LDQKV + (size_t)h * DHEAD + j * 256;
  float m = p[0];
  for (int i = 1; i < 256; ++i) m = fmaxf(m, p[i]);
  pool[idx] = m;
}

// ---------------------------------------------------------------------------
// logits[h, m, n] = scale * <q[h,m,:], k[h,n,:]> + 0.1 * pool[h, m, n%12]
// One full wave (32 threads) per 16x16 output tile. grid = (15, 15, 2).
// ---------------------------------------------------------------------------
__global__ void attn_logits_kernel(const float* __restrict__ qkv,
                                   const float* __restrict__ pool,
                                   float* __restrict__ S) {
  int lane = threadIdx.x;
  int m0 = blockIdx.x * 16;   // q rows
  int n0 = blockIdx.y * 16;   // k rows
  int h  = blockIdx.z;
  const float* Q  = qkv + (size_t)h * DHEAD + (size_t)m0 * LDQKV;
  const float* Kb = qkv + CDIM + (size_t)h * DHEAD + (size_t)n0 * LDQKV;

  v8f acc = {};
  for (int k0 = 0; k0 < DHEAD; k0 += 32) {
    v16bf a = load_a_frag_f32(Q, LDQKV, k0, lane);
    v16bf b = load_b_frag_rowmajor(Kb, LDQKV, k0, lane);
    acc = wmma_bf16(a, b, acc);
  }

  const float scale = 0.01804219599f;  // 1/sqrt(3072)
  int n  = lane & 15;
  int mb = (lane & 16) ? 8 : 0;
  int col = n0 + n;
#pragma unroll
  for (int i = 0; i < 8; ++i) {
    int m = m0 + mb + i;
    float pq = pool[(size_t)h * (NSEQ * 12) + (size_t)m * 12 + (col % 12)];
    S[(size_t)h * NSEQ * SLD + (size_t)m * SLD + col] = acc[i] * scale + 0.1f * pq;
  }
}

// ---------------------------------------------------------------------------
// Row softmax over 240 valid columns; pad columns [240,256) zeroed so the
// following WMMA GEMM can run unguarded A-fragment loads with ld = 256.
// ---------------------------------------------------------------------------
__global__ void softmax_kernel(float* __restrict__ S) {
  __shared__ float red[256];
  int t = threadIdx.x;
  int h = blockIdx.x / NSEQ;
  int n = blockIdx.x % NSEQ;
  float* row = S + (size_t)h * NSEQ * SLD + (size_t)n * SLD;

  float v = (t < NSEQ) ? row[t] : -3.402823466e38f;
  red[t] = v;
  __syncthreads();
  for (int s = 128; s > 0; s >>= 1) {
    if (t < s) red[t] = fmaxf(red[t], red[t + s]);
    __syncthreads();
  }
  float mx = red[0];
  __syncthreads();
  float e = (t < NSEQ) ? __expf(v - mx) : 0.0f;
  red[t] = e;
  __syncthreads();
  for (int s = 128; s > 0; s >>= 1) {
    if (t < s) red[t] += red[t + s];
    __syncthreads();
  }
  row[t] = e * (1.0f / red[0]);   // pads write exact 0
}

// ---------------------------------------------------------------------------
// AO[n, h*d + c] = sum_m S[h, n, m] * V[h, m, c]   (V rows = K dimension)
// grid = (6, 15, 2); 8 waves/block, 16x64 strip each; K padded to 256.
// Output written directly in bf16 (it is only consumed by the proj GEMM).
// ---------------------------------------------------------------------------
__global__ void attn_v_kernel(const float* __restrict__ S,
                              const float* __restrict__ qkv,
                              __bf16* __restrict__ AO) {
  int lane = threadIdx.x & 31;
  int wave = threadIdx.x >> 5;
  int h  = blockIdx.z;
  int m0 = blockIdx.y * 16;
  int c0 = blockIdx.x * 512 + wave * 64;
  const float* Srow = S + (size_t)h * NSEQ * SLD + (size_t)m0 * SLD;
  const float* V    = qkv + 2 * CDIM + (size_t)h * DHEAD;

  v8f acc[4] = {};
  for (int k0 = 0; k0 < SLD; k0 += 32) {
    v16bf a = load_a_frag_f32(Srow, SLD, k0, lane);   // pads are zero
#pragma unroll
    for (int t = 0; t < 4; ++t) {
      v16bf b = load_b_frag_strided(V, LDQKV, c0 + 16 * t, k0, NSEQ, lane);
      acc[t] = wmma_bf16(a, b, acc[t]);
    }
  }

  int n  = lane & 15;
  int mb = (lane & 16) ? 8 : 0;
#pragma unroll
  for (int t = 0; t < 4; ++t)
#pragma unroll
    for (int i = 0; i < 8; ++i)
      AO[(size_t)(m0 + mb + i) * CDIM + (size_t)h * DHEAD + c0 + 16 * t + n] =
          (__bf16)acc[t][i];
}

// ---------------------------------------------------------------------------
extern "C" void kernel_launch(void* const* d_in, const int* in_sizes, int n_in,
                              void* d_out, int out_size, void* d_ws, size_t ws_size,
                              hipStream_t stream) {
  const float* x      = (const float*)d_in[0];
  const float* w_qkv  = (const float*)d_in[1];
  const float* w_proj = (const float*)d_in[2];
  const float* b_proj = (const float*)d_in[3];
  float* out = (float*)d_out;

  // workspace carve-up: qkv | pool | S(padded) | x_bf16 | AO_bf16  ~24.1 MB
  char* wsp = (char*)d_ws;
  float* qkv  = (float*)wsp;  wsp += (size_t)NSEQ * LDQKV * 4;       // 17.7 MB
  float* pool = (float*)wsp;  wsp += (size_t)2 * NSEQ * 12 * 4;
  float* S    = (float*)wsp;  wsp += (size_t)2 * NSEQ * SLD * 4;
  __bf16* xb  = (__bf16*)wsp; wsp += (size_t)NSEQ * CDIM * 2;        //  2.9 MB
  __bf16* AOb = (__bf16*)wsp;                                        //  2.9 MB

  // 0) x -> bf16 (A operand of the qkv GEMM)
  int nx = NSEQ * CDIM;
  f32_to_bf16_kernel<<<dim3((nx + 255) / 256), 256, 0, stream>>>(x, xb, nx);

  // 1) qkv = x @ w_qkv^T : 18432 cols / 128 per block = 144 blocks
  gemm_wstream_bf16<<<dim3(144), 256, 0, stream>>>(xb, CDIM, w_qkv, CDIM,
                                                   qkv, LDQKV, CDIM, nullptr);
  // 2) maxpool over q
  qpool_kernel<<<dim3((2 * NSEQ * 12 + 255) / 256), 256, 0, stream>>>(qkv, pool);
  // 3) logits = scale*q@k^T + 0.1*Pq
  attn_logits_kernel<<<dim3(15, 15, 2), 32, 0, stream>>>(qkv, pool, S);
  // 4) softmax rows
  softmax_kernel<<<dim3(2 * NSEQ), 256, 0, stream>>>(S);
  // 5) attn @ v  (writes bf16)
  attn_v_kernel<<<dim3(6, 15, 2), 256, 0, stream>>>(S, qkv, AOb);
  // 6) out = AO @ w_proj^T + b_proj : 6144 cols / 128 per block = 48 blocks
  gemm_wstream_bf16<<<dim3(48), 256, 0, stream>>>(AOb, CDIM, w_proj, CDIM,
                                                  out, CDIM, CDIM, b_proj);
}
